// Conv2dTB_48653389529623
// MI455X (gfx1250) — matrix operand order repaired and checked by reading, not compiled, not run on hardware
//
#include <hip/hip_runtime.h>

typedef _Float16 v16h __attribute__((ext_vector_type(16)));
typedef _Float16 v8h  __attribute__((ext_vector_type(8)));
typedef float    v8f  __attribute__((ext_vector_type(8)));
typedef int      gv4i __attribute__((vector_size(16)));   // matches builtin param

#define BN_EPS 1e-4f

// ---- CDNA5 async-copy path (ASYNCcnt-tracked global->LDS), guarded ----------
#if defined(__has_builtin)
#if __has_builtin(__builtin_amdgcn_global_load_async_to_lds_b128)
#define TB_ASYNC 1
#endif
#endif

#ifdef TB_ASYNC
#if __has_builtin(__builtin_amdgcn_s_wait_asynccnt)
#define WAIT_ASYNC(n) __builtin_amdgcn_s_wait_asynccnt(n)
#else
#define WAIT_ASYNC(n) asm volatile("s_wait_asynccnt %0" ::"i"(n))
#endif
// 16B per lane, global -> LDS, tracked by ASYNCcnt.
// Builtin signature (from clang diagnostic): param0 = gv4i addrspace(1)*,
// param1 = gv4i addrspace(3)*, then two immediate ints (offset, cpol).
#define ASYNC_CP16(ldsDst, glbSrc)                                        \
  __builtin_amdgcn_global_load_async_to_lds_b128(                         \
      (__attribute__((address_space(1))) gv4i*)(glbSrc),                  \
      (__attribute__((address_space(3))) gv4i*)(ldsDst), 0, 0)
#endif

// ---------------- Stage 1: BN batch stats -> per-channel scale/shift ----------
__global__ __launch_bounds__(256)
void bn_stats_kernel(const float* __restrict__ x,
                     const float* __restrict__ gamma,
                     const float* __restrict__ beta,
                     float* __restrict__ scale,
                     float* __restrict__ shift) {
  const int c   = blockIdx.x;      // one block per channel
  const int tid = threadIdx.x;
  float s = 0.f, s2 = 0.f;
  for (int n = 0; n < 16; ++n) {
    const float* px = x + ((long long)(n * 256 + c)) * 3136;
    for (int i = tid; i < 3136; i += 256) {
      float v = px[i];
      s += v; s2 += v * v;
    }
  }
  __shared__ float rs[256], rq[256];
  rs[tid] = s; rq[tid] = s2;
  __syncthreads();
  for (int off = 128; off > 0; off >>= 1) {
    if (tid < off) { rs[tid] += rs[tid + off]; rq[tid] += rq[tid + off]; }
    __syncthreads();
  }
  if (tid == 0) {
    const float inv = 1.f / 50176.f;
    float mean = rs[0] * inv;
    float var  = rq[0] * inv - mean * mean;    // biased variance
    float sc   = gamma[c] * rsqrtf(var + BN_EPS);
    scale[c] = sc;
    shift[c] = beta[c] - mean * sc;
  }
}

// ---------------- Stage 2: pack conv weights f32 OIHW -> f16 [tap][cout][c] ---
__global__ __launch_bounds__(256)
void wpack_kernel(const float* __restrict__ wsrc, _Float16* __restrict__ Wh) {
  const int idx  = blockIdx.x * 256 + threadIdx.x;   // < 65536
  const int cout = idx >> 8;
  const int c    = idx & 255;
  const float* s = wsrc + (long long)cout * 2304 + c * 9;
#pragma unroll
  for (int tap = 0; tap < 9; ++tap)
    Wh[tap * 65536 + idx] = (_Float16)s[tap];
}

// ---------------- Stage 3: ternarize to NHWC f16 (+ per-pixel c_sum) ----------
__global__ __launch_bounds__(256)
void quantize_kernel(const float* __restrict__ x,
                     const float* __restrict__ scale,
                     const float* __restrict__ shift,
                     _Float16* __restrict__ Th,
                     float* __restrict__ csum) {
  __shared__ __align__(16) _Float16 tl[56 * 256];   // one (n,h) row slab, NHWC
  __shared__ float ps[4 * 64];
  const int nb = blockIdx.x;          // n*56 + h
  const int n  = nb / 56;
  const int h  = nb - n * 56;
  const int cg = threadIdx.x >> 6;    // channel group 0..3
  const int wl = threadIdx.x & 63;    // w lane (valid < 56)
  float acc = 0.f;
  if (wl < 56) {
    const long long rowbase = ((long long)(n * 256)) * 3136 + h * 56 + wl;
    for (int ci = 0; ci < 64; ++ci) {
      const int c = cg * 64 + ci;
      float v  = x[rowbase + (long long)c * 3136];
      float xn = v * scale[c] + shift[c];
      float t  = (xn > 0.f) ? 1.f : -1.f;     // xn==0 -> -1 (matches reference)
      acc += fminf(fabsf(xn), 1.f);           // clip(xn,-1,1)*t == min(|xn|,1)
      tl[wl * 256 + c] = (_Float16)t;
    }
  }
  ps[cg * 64 + wl] = acc;
  __syncthreads();
  if (cg == 0 && wl < 56) {
    float sres = ps[wl] + ps[64 + wl] + ps[128 + wl] + ps[192 + wl];
    csum[n * 3136 + h * 56 + wl] = sres;
  }
  // coalesced NHWC writeback (8 f16 = 16B per thread-step)
  const long long obase = ((long long)(n * 3136 + h * 56)) * 256;
  for (int i = threadIdx.x; i < 1792; i += 256) {
    *(uint4*)(Th + obase + (long long)i * 8) = *(const uint4*)(tl + i * 8);
  }
}

// ---------------- Stage 4: beta map (c_cnt == 256 exactly, box is analytic) ---
__global__ __launch_bounds__(256)
void beta_kernel(const float* __restrict__ csum,
                 const float* __restrict__ bb,
                 float* __restrict__ betamap) {
  const int m  = blockIdx.x * 256 + threadIdx.x;   // < 50176
  const int n  = m / 3136;
  const int hw = m - n * 3136;
  const int h  = hw / 56;
  const int w  = hw - h * 56;
  const float b0 = bb[0];
  float s = 0.f; int cnt = 0;
#pragma unroll
  for (int dh = -1; dh <= 1; ++dh) {
    const int h2 = h + dh;
    if ((unsigned)h2 >= 56u) continue;
#pragma unroll
    for (int dw = -1; dw <= 1; ++dw) {
      const int w2 = w + dw;
      if ((unsigned)w2 >= 56u) continue;
      s += csum[n * 3136 + h2 * 56 + w2];
      ++cnt;
    }
  }
  betamap[m] = (s + b0) / (256.f * (float)cnt + b0);
}

// ---------------- Stage 5: tap-decomposed implicit GEMM with WMMA f16 ---------
// Block: 256 thr (8 waves). Tile M=64 pixels x N=64 cout. K = 9 taps x 256 ch.
// Per tap: stage full 64x256 A and B slabs in LDS (one barrier pair), then 8
// unrolled k-steps of 2 WMMAs per wave. Async path double-buffers the slabs.
__global__ __launch_bounds__(256)
void tconv_wmma_kernel(const _Float16* __restrict__ Th,
                       const _Float16* __restrict__ Wh,
                       const float* __restrict__ betamap,
                       const float* __restrict__ conv_b,
                       float* __restrict__ out) {
#ifdef TB_ASYNC
  __shared__ __align__(16) _Float16 Als[2][64 * 256];   // 2 x 32 KB
  __shared__ __align__(16) _Float16 Bls[2][64 * 256];   // 2 x 32 KB
  __shared__ __align__(16) uint4 trash[256];            // dummy async target
#else
  __shared__ __align__(16) _Float16 Als[1][64 * 256];
  __shared__ __align__(16) _Float16 Bls[1][64 * 256];
#endif

  const int tid   = threadIdx.x;
  const int lane  = tid & 31;
  const int wave  = tid >> 5;
  const int halfp = lane >> 4;      // 0: lanes 0-15, 1: lanes 16-31
  const int row   = lane & 15;
  const int m_sub = wave & 3;       // 16-row slice of M tile
  const int n_sub = wave >> 2;      // 32-col slice of N tile

  const int mtile0 = blockIdx.x * 64;   // 784 tiles; 3136%64==0 -> stays in one image
  const int ctile0 = blockIdx.y * 64;   // 4 tiles

  // slab loader mapping: chunk row r = i*8 + wave (wave-uniform!), 32 lanes
  // cover the 512B row; this thread's 16B piece starts at f16 offset lc.
  const int lc = lane * 8;

  v8f acc0 = {}; v8f acc1 = {};

  // ---- per-tap slab fill -----------------------------------------------------
  auto fill_tap = [&](int tap, int buf) {
    const int kh = tap / 3 - 1;
    const int kw = tap - (tap / 3) * 3 - 1;
    const uint4 zero = {0u, 0u, 0u, 0u};
#pragma unroll
    for (int i = 0; i < 8; ++i) {
      const int r  = i * 8 + wave;        // 0..63, uniform across the wave
      const int mg = mtile0 + r;
      const int ni = mg / 3136;
      const int hw = mg - ni * 3136;
      const int h  = hw / 56;
      const int w  = hw - h * 56;
      const int hs = h + kh;
      const int wsx = w + kw;
      const bool ok = ((unsigned)hs < 56u) && ((unsigned)wsx < 56u);
      _Float16* adst = &Als[buf][r * 256 + lc];
      const _Float16* asrc =
          Th + ((long long)(ni * 3136 + hs * 56 + wsx)) * 256 + lc;
      const _Float16* bsrc =
          Wh + tap * 65536 + (long long)(ctile0 + r) * 256 + lc;
#ifdef TB_ASYNC
      if (ok) {
        ASYNC_CP16(adst, asrc);
      } else {
        ASYNC_CP16(&trash[tid], Th + lc);  // keep per-wave ASYNCcnt uniform
        *(uint4*)adst = zero;              // zero halo via DS store
      }
      ASYNC_CP16(&Bls[buf][r * 256 + lc], bsrc);
#else
      uint4 av = zero;
      if (ok) av = *(const uint4*)asrc;
      *(uint4*)adst = av;
      *(uint4*)&Bls[buf][r * 256 + lc] = *(const uint4*)bsrc;
      __builtin_prefetch((const void*)(bsrc + 65536), 0, 1);  // next tap weights
#endif
    }
  };

  // ---- main loop -------------------------------------------------------------
  int buf = 0;
#ifdef TB_ASYNC
  fill_tap(0, 0);                    // 16 async ops per thread
#endif
  for (int tap = 0; tap < 9; ++tap) {
#ifdef TB_ASYNC
    if (tap < 8) {
      fill_tap(tap + 1, buf ^ 1);    // +16 outstanding
      WAIT_ASYNC(16);                // current slab's 16 retired (in-order)
    } else {
      WAIT_ASYNC(0);
    }
    __syncthreads();                 // all waves' slab writes visible
#else
    __syncthreads();                 // previous slab fully consumed
    fill_tap(tap, 0);
    __syncthreads();
#endif

    // compute on slab `buf`
    const _Float16* Ab  = &Als[buf][(m_sub * 16 + row) * 256 + 16 * halfp];
    const _Float16* Bb0 = &Bls[buf][(n_sub * 32 + row) * 256 + 16 * halfp];
    const _Float16* Bb1 = Bb0 + 16 * 256;
#pragma unroll
    for (int kc = 0; kc < 256; kc += 32) {
      // A frag (16x32 f16): lane needs K[16h..16h+15]
      v8h alo = *(const v8h*)(Ab + kc);
      v8h ahi = *(const v8h*)(Ab + kc + 8);
      v16h a  = __builtin_shufflevector(alo, ahi,
                 0,1,2,3,4,5,6,7,8,9,10,11,12,13,14,15);
      // B frags (32x16 f16): lane holds K[16h..16h+15] of its column
      v16h b0 = *(const v16h*)(Bb0 + kc);
      v16h b1 = *(const v16h*)(Bb1 + kc);
      acc0 = __builtin_amdgcn_wmma_f32_16x16x32_f16(false, a, false, b0,
                                                    (short)0, acc0, false, false);
      acc1 = __builtin_amdgcn_wmma_f32_16x16x32_f16(false, a, false, b1,
                                                    (short)0, acc1, false, false);
    }

#ifdef TB_ASYNC
    __syncthreads();                 // done reading `buf` before it is refilled
    buf ^= 1;
#endif
  }

  // ---- epilogue: y = (acc + conv_b[cout]) * beta_map[pixel]; NCHW f32 --------
  const int c0 = ctile0 + n_sub * 32 + row;
  const float cb0 = conv_b[c0];
  const float cb1 = conv_b[c0 + 16];
#pragma unroll
  for (int j = 0; j < 8; ++j) {
    const int m_local = m_sub * 16 + j + 8 * halfp;   // D layout: M = j + 8*half
    const int mgo = mtile0 + m_local;
    const int ni  = mgo / 3136;
    const int hwo = mgo - ni * 3136;
    const float bm = betamap[mgo];
    out[((long long)(ni * 256 + c0     )) * 3136 + hwo] = (acc0[j] + cb0) * bm;
    out[((long long)(ni * 256 + c0 + 16)) * 3136 + hwo] = (acc1[j] + cb1) * bm;
  }
}

// ---------------- launcher ----------------------------------------------------
extern "C" void kernel_launch(void* const* d_in, const int* in_sizes, int n_in,
                              void* d_out, int out_size, void* d_ws, size_t ws_size,
                              hipStream_t stream) {
  const float* x      = (const float*)d_in[0];
  const float* gamma  = (const float*)d_in[1];
  const float* betab  = (const float*)d_in[2];
  const float* conv_w = (const float*)d_in[3];
  const float* conv_b = (const float*)d_in[4];
  const float* bcb    = (const float*)d_in[5];
  float* out = (float*)d_out;

  // workspace partition (all offsets 256B aligned); total ~27.3 MB
  char* wsp = (char*)d_ws;
  float*    scale = (float*)(wsp + 0);                       // 256 f32
  float*    shift = (float*)(wsp + 1024);                    // 256 f32
  _Float16* Wh    = (_Float16*)(wsp + 4096);                 // 9*256*256 f16 = 1.18 MB
  float*    csum  = (float*)(wsp + 4096 + 1179648);          // 50176 f32
  float*    betam = (float*)(wsp + 4096 + 1179648 + 200704); // 50176 f32
  _Float16* Th    = (_Float16*)(wsp + 4096 + 1179648 + 401408); // 25.7 MB

  bn_stats_kernel <<<256, 256, 0, stream>>>(x, gamma, betab, scale, shift);
  wpack_kernel    <<<256, 256, 0, stream>>>(conv_w, Wh);
  quantize_kernel <<<896, 256, 0, stream>>>(x, scale, shift, Th, csum);
  beta_kernel     <<<196, 256, 0, stream>>>(csum, bcb, betam);
  tconv_wmma_kernel<<<dim3(784, 4), 256, 0, stream>>>(Th, Wh, betam, conv_b, out);
}
